// Model_25950192402800
// MI455X (gfx1250) — compile-verified, matrix operand order checked by reading, not run on hardware
//
#include <hip/hip_runtime.h>
#include <stdint.h>

// ---------------- problem constants ----------------
#define BATCH   512
#define SEQ     256
#define DIN     64
#define HID     512
#define G4      2048   // 4*HID
#define PRED    24

// ---------------- workspace layout (bytes, 256-aligned) ----------------
#define OFF_GFRAG   0u         // gates weight fragments: 18 kt x 128 nt x 512 ushort
#define OFF_UFRAG   2359296u   // Ua fragments: 32 x 4 x 512 ushort
#define OFF_WAFRAG  2490368u   // Wa fragments: 2 x 4 x 512 ushort
#define OFF_VAFRAG  2498560u   // Va fragments: 2 x 4 x 512 ushort
#define OFF_XATT    2506752u   // x_att bf16 [512,64]
#define OFF_HBF     2572288u   // h bf16 [512,512]
#define OFF_CBF     3096576u   // c bf16 [512,512]
#define OFF_HF      3620864u   // h f32  [512,512]
#define OFF_CF      4669440u   // c f32  [512,512]
#define OFF_GATES   5718016u   // gates f32 [512,2048]
// total ~9.45 MB

typedef __attribute__((ext_vector_type(16))) __bf16 bf16x16;
typedef __attribute__((ext_vector_type(8)))  float  f32x8;
typedef int i32x4 __attribute__((vector_size(16)));

#if __has_builtin(__builtin_amdgcn_global_load_async_to_lds_b128)
#define HAVE_ASYNC_LDS 1
#else
#define HAVE_ASYNC_LDS 0
#endif

__device__ __forceinline__ void wait_async_copies() {
#if HAVE_ASYNC_LDS
#if __has_builtin(__builtin_amdgcn_s_wait_asynccnt)
    __builtin_amdgcn_s_wait_asynccnt(0);
#else
    asm volatile("s_wait_asynccnt 0" ::: "memory");
#endif
#endif
}

// ---------------- scalar bf16 helpers (RNE) ----------------
__device__ __forceinline__ unsigned short f2bf(float f) {
    union { float f; unsigned u; } a; a.f = f;
    unsigned r = a.u + 0x7FFFu + ((a.u >> 16) & 1u);
    return (unsigned short)(r >> 16);
}
__device__ __forceinline__ __bf16 bfbits(unsigned short u) {
    union { unsigned short u; __bf16 h; } x; x.u = u; return x.h;
}

// K index inside a 16x32 bf16 A fragment (CDNA5 ISA 7.12.2 layout)
__device__ __forceinline__ int a_kidx(int lane, int e) {
    return ((e & 8) << 1) + ((lane >> 4) << 3) + (e & 7);
}

// ---------------- WMMA wrapper ----------------
__device__ __forceinline__ f32x8 wmma_bf16(f32x8 acc, bf16x16 a, bf16x16 b) {
    return __builtin_amdgcn_wmma_f32_16x16x32_bf16(
        false, a, false, b, (short)0, acc, false, false);
}

// ---------------- fragment loaders ----------------
// A fragment from row-major bf16 matrix (row stride ld elements)
__device__ __forceinline__ bf16x16 load_a_bf16(const unsigned short* base, int ld,
                                               int mbase, int kb) {
    const int lane = threadIdx.x & 31;
    const unsigned short* p = base + (size_t)(mbase + (lane & 15)) * ld
                                   + kb + ((lane >> 4) << 3);
    union U { uint4 v; __bf16 h[8]; } u0, u1;
    u0.v = *reinterpret_cast<const uint4*>(p);
    u1.v = *reinterpret_cast<const uint4*>(p + 16);
    bf16x16 r;
#pragma unroll
    for (int i = 0; i < 8; ++i) { r[i] = u0.h[i]; r[8 + i] = u1.h[i]; }
    return r;
}
// A fragment from row-major f32 matrix (converted on the fly)
__device__ __forceinline__ bf16x16 load_a_f32(const float* base, size_t ld,
                                              int mbase, int kb) {
    const int lane = threadIdx.x & 31;
    const float* p = base + (size_t)(mbase + (lane & 15)) * ld
                          + kb + ((lane >> 4) << 3);
    float4 q0 = *reinterpret_cast<const float4*>(p);
    float4 q1 = *reinterpret_cast<const float4*>(p + 4);
    float4 q2 = *reinterpret_cast<const float4*>(p + 16);
    float4 q3 = *reinterpret_cast<const float4*>(p + 20);
    float f[16] = { q0.x,q0.y,q0.z,q0.w, q1.x,q1.y,q1.z,q1.w,
                    q2.x,q2.y,q2.z,q2.w, q3.x,q3.y,q3.z,q3.w };
    bf16x16 r;
#pragma unroll
    for (int i = 0; i < 16; ++i) r[i] = bfbits(f2bf(f[i]));
    return r;
}
// A fragment from LDS row-major bf16
__device__ __forceinline__ bf16x16 load_a_lds(const unsigned short* s, int ld, int kb) {
    const int lane = threadIdx.x & 31;
    const unsigned short* p = s + (lane & 15) * ld + kb + ((lane >> 4) << 3);
    union U { uint4 v; __bf16 h[8]; } u0, u1;
    u0.v = *reinterpret_cast<const uint4*>(p);
    u1.v = *reinterpret_cast<const uint4*>(p + 16);
    bf16x16 r;
#pragma unroll
    for (int i = 0; i < 8; ++i) { r[i] = u0.h[i]; r[8 + i] = u1.h[i]; }
    return r;
}
// B fragment from pre-swizzled fragment-major buffer (32 lanes x 16 bf16)
// works for both global and LDS pointers
__device__ __forceinline__ bf16x16 load_b_frag(const unsigned short* F) {
    const int lane = threadIdx.x & 31;
    const uint4* p = reinterpret_cast<const uint4*>(F + lane * 16);
    union U { uint4 v[2]; bf16x16 r; } u;
    u.v[0] = p[0]; u.v[1] = p[1];
    return u.r;
}

// ================= prep: build bf16 pre-swizzled weight fragments =================
// fragment ids: [0,2304) gates (18 kt x 128 nt), [2304,2432) Ua (32x4),
//               [2432,2440) Wa (2x4), [2440,2448) Va (2x4)
__global__ void prep_weights(const float* __restrict__ Wa, const float* __restrict__ Ua,
                             const float* __restrict__ Va, const float* __restrict__ Wih,
                             const float* __restrict__ Whh, unsigned short* __restrict__ gfrag,
                             unsigned short* __restrict__ ufrag, unsigned short* __restrict__ wafrag,
                             unsigned short* __restrict__ vafrag) {
    int tid = blockIdx.x * blockDim.x + threadIdx.x;
    if (tid >= 2448 * 512) return;
    int fid  = tid >> 9;
    int li   = tid & 511;
    int lane = li >> 4, e = li & 15;
    int kidx = a_kidx(lane, e);
    int col  = lane & 15;
    if (fid < 2304) {
        int kt = fid / 128, nt = fid % 128;
        int n = nt * 16 + col, k = kt * 32 + kidx;
        float v = (k < 64) ? Wih[(size_t)n * 64 + k] : Whh[(size_t)n * 512 + (k - 64)];
        gfrag[(size_t)fid * 512 + li] = f2bf(v);
    } else if (fid < 2432) {
        int u = fid - 2304; int kt = u / 4, nt = u % 4;
        int k = kt * 32 + kidx, n = nt * 16 + col;
        ufrag[(size_t)u * 512 + li] = f2bf(Ua[(size_t)k * 64 + n]);
    } else if (fid < 2440) {
        int w = fid - 2432; int kt = w / 4, nt = w % 4;
        wafrag[(size_t)w * 512 + li] = f2bf(Wa[(size_t)(kt * 32 + kidx) * 64 + nt * 16 + col]);
    } else {
        int w = fid - 2440; int kt = w / 4, nt = w % 4;
        vafrag[(size_t)w * 512 + li] = f2bf(Va[(size_t)(kt * 32 + kidx) * 64 + nt * 16 + col]);
    }
}

__global__ void init_state(float* hf, float* cf, unsigned short* hb, unsigned short* cb) {
    int i = blockIdx.x * blockDim.x + threadIdx.x;
    if (i < BATCH * HID) { hf[i] = 0.f; cf[i] = 0.f; hb[i] = 0; cb[i] = 0; }
}

// ================= per-step: attention gate =================
// block = 128 threads (4 waves), grid = 32 (16 batch rows per block)
__global__ void attn_step(const float* __restrict__ x, const float* __restrict__ ba,
                          const unsigned short* __restrict__ wafrag,
                          const unsigned short* __restrict__ ufrag,
                          const unsigned short* __restrict__ vafrag,
                          const unsigned short* __restrict__ hbf,
                          const unsigned short* __restrict__ cbf,
                          unsigned short* __restrict__ xatt, int t) {
    __shared__ unsigned short s_lds[16 * 64];
    __shared__ float a_lds[16 * 64];
    __shared__ float rmax[16], rsum[16];

    const int mbase = blockIdx.x * 16;
    const int w     = threadIdx.x >> 5;      // n-tile 0..3
    const int lane  = threadIdx.x & 31;
    const float* xt = x + (size_t)t * DIN;   // row b at stride SEQ*DIN

    f32x8 acc = {};
    // a1 = x_t @ Wa   (K = 64)
#pragma unroll
    for (int kt = 0; kt < 2; ++kt) {
        bf16x16 a = load_a_f32(xt, (size_t)SEQ * DIN, mbase, kt * 32);
        bf16x16 b = load_b_frag(wafrag + (size_t)(kt * 4 + w) * 512);
        acc = wmma_bf16(acc, a, b);
    }
    // a2 = [h|c] @ Ua (K = 1024)
    for (int kt = 0; kt < 32; ++kt) {
        bf16x16 a = (kt < 16) ? load_a_bf16(hbf, HID, mbase, kt * 32)
                              : load_a_bf16(cbf, HID, mbase, (kt - 16) * 32);
        bf16x16 b = load_b_frag(ufrag + (size_t)(kt * 4 + w) * 512);
        acc = wmma_bf16(acc, a, b);
    }
    // s = tanh(acc + ba)  -> LDS (bf16)
    {
        int col = w * 16 + (lane & 15);
        float bias = ba[col];
        int mh = (lane >> 4) << 3;
#pragma unroll
        for (int r = 0; r < 8; ++r)
            s_lds[(r + mh) * 64 + col] = f2bf(tanhf(acc[r] + bias));
    }
    __syncthreads();
    // a = s @ Va (K = 64), A from LDS
    f32x8 acc2 = {};
#pragma unroll
    for (int kt = 0; kt < 2; ++kt) {
        bf16x16 a = load_a_lds(s_lds, 64, kt * 32);
        bf16x16 b = load_b_frag(vafrag + (size_t)(kt * 4 + w) * 512);
        acc2 = wmma_bf16(acc2, a, b);
    }
    {
        int col = w * 16 + (lane & 15);
        int mh = (lane >> 4) << 3;
#pragma unroll
        for (int r = 0; r < 8; ++r)
            a_lds[(r + mh) * 64 + col] = acc2[r];
    }
    __syncthreads();
    // row softmax stats (16 rows, 64 cols)
    if (threadIdx.x < 16) {
        int row = threadIdx.x;
        float mx = -3.0e38f;
        for (int cIt = 0; cIt < 64; ++cIt) mx = fmaxf(mx, a_lds[row * 64 + cIt]);
        float sm = 0.f;
        for (int cIt = 0; cIt < 64; ++cIt) sm += __expf(a_lds[row * 64 + cIt] - mx);
        rmax[row] = mx; rsum[row] = sm;
    }
    __syncthreads();
    // x_att = softmax(a) * x_t  -> bf16 global
    for (int i = 0; i < 8; ++i) {
        int idx = threadIdx.x * 8 + i;
        int row = idx >> 6, col = idx & 63;
        float alpha = __expf(a_lds[row * 64 + col] - rmax[row]) / rsum[row];
        float xv = x[(size_t)(mbase + row) * SEQ * DIN + (size_t)t * DIN + col];
        xatt[(size_t)(mbase + row) * 64 + col] = f2bf(alpha * xv);
    }
}

// ================= per-step: gates GEMM [512,576]x[576,2048] =================
// block = 256 (8 waves), grid = 128; each wave: one 16(M) x 64(N) strip.
// All 8 waves of a block share ns -> share B fragments; stage them through LDS
// with double-buffered async global->LDS copies (ASYNCcnt / s_wait_asynccnt).
__global__ void gates_gemm(const unsigned short* __restrict__ xatt,
                           const unsigned short* __restrict__ hbf,
                           const unsigned short* __restrict__ gfrag,
                           float* __restrict__ gates) {
    __shared__ unsigned short bstage[2][4 * 512];   // 2 x 4KB

    const int gw   = blockIdx.x * 8 + (threadIdx.x >> 5);
    const int mt   = gw & 31, ns = gw >> 5;         // ns uniform per block
    const int lane = threadIdx.x & 31;
    const int mbase = mt * 16, nbase = ns * 64;

    // stage k-tile kt's 4 B fragments (4KB contiguous) into bstage[buf]:
    // each of 256 threads copies 16 bytes, layout-preserving.
    auto stage = [&](int kt, int buf) {
        const unsigned short* g = gfrag + ((size_t)kt * 128 + ns * 4) * 512
                                        + threadIdx.x * 8;
        unsigned short* l = &bstage[buf][threadIdx.x * 8];
#if HAVE_ASYNC_LDS
        __builtin_amdgcn_global_load_async_to_lds_b128(
            (__attribute__((address_space(1))) i32x4*)g,
            (__attribute__((address_space(3))) i32x4*)l, 0, 0);
#else
        *reinterpret_cast<uint4*>(l) = *reinterpret_cast<const uint4*>(g);
#endif
    };

    stage(0, 0);

    f32x8 acc0 = {}, acc1 = {}, acc2 = {}, acc3 = {};
    for (int kt = 0; kt < 18; ++kt) {
        const int cur = kt & 1;
        wait_async_copies();      // own async copies into bstage[cur] done
        __syncthreads();          // everyone's copies visible; prior reads done
        if (kt + 1 < 18) {
            stage(kt + 1, 1 - cur);
            // prefetch next A fragment row region
            const unsigned short* an = (kt + 1 < 2)
                ? xatt + (size_t)(mbase + (lane & 15)) * 64 + (kt + 1) * 32
                : hbf  + (size_t)(mbase + (lane & 15)) * HID + (kt - 1) * 32;
            __builtin_prefetch(an, 0, 1);
        }
        bf16x16 a = (kt < 2) ? load_a_bf16(xatt, 64, mbase, kt * 32)
                             : load_a_bf16(hbf, HID, mbase, (kt - 2) * 32);
        const unsigned short* fb = &bstage[cur][0];
        acc0 = wmma_bf16(acc0, a, load_b_frag(fb));
        acc1 = wmma_bf16(acc1, a, load_b_frag(fb + 512));
        acc2 = wmma_bf16(acc2, a, load_b_frag(fb + 1024));
        acc3 = wmma_bf16(acc3, a, load_b_frag(fb + 1536));
    }
    const int mh = (lane >> 4) << 3, cl = lane & 15;
#pragma unroll
    for (int r = 0; r < 8; ++r) {
        size_t row = (size_t)(mbase + r + mh) * G4 + nbase + cl;
        gates[row +  0] = acc0[r];
        gates[row + 16] = acc1[r];
        gates[row + 32] = acc2[r];
        gates[row + 48] = acc3[r];
    }
}

// ================= per-step: LSTM cell update =================
__global__ void cell_update(const float* __restrict__ gates,
                            const float* __restrict__ bih, const float* __restrict__ bhh,
                            float* __restrict__ hf, float* __restrict__ cf,
                            unsigned short* __restrict__ hb, unsigned short* __restrict__ cb) {
    int idx = blockIdx.x * blockDim.x + threadIdx.x;
    if (idx >= BATCH * HID) return;
    int b = idx >> 9, j = idx & 511;
    size_t base = (size_t)b * G4;
    float gi = gates[base + j]          + bih[j]          + bhh[j];
    float gf = gates[base + 512 + j]    + bih[512 + j]    + bhh[512 + j];
    float gg = gates[base + 1024 + j]   + bih[1024 + j]   + bhh[1024 + j];
    float go = gates[base + 1536 + j]   + bih[1536 + j]   + bhh[1536 + j];
    float si = 1.f / (1.f + __expf(-gi));
    float sf = 1.f / (1.f + __expf(-gf));
    float so = 1.f / (1.f + __expf(-go));
    float cn = sf * cf[idx] + si * tanhf(gg);
    float hn = so * tanhf(cn);
    cf[idx] = cn; hf[idx] = hn;
    cb[idx] = f2bf(cn); hb[idx] = f2bf(hn);
}

// ================= final FC: y = h @ fc_w.T + fc_b =================
__global__ void final_fc(const float* __restrict__ hf, const float* __restrict__ fcw,
                         const float* __restrict__ fcb, float* __restrict__ out) {
    int idx = blockIdx.x * blockDim.x + threadIdx.x;
    if (idx >= BATCH * PRED) return;
    int b = idx / PRED, o = idx % PRED;
    float s = fcb[o];
    const float* hr = hf + (size_t)b * HID;
    const float* wr = fcw + (size_t)o * HID;
    for (int k = 0; k < HID; ++k) s += hr[k] * wr[k];
    out[idx] = s;
}

// ================= host =================
extern "C" void kernel_launch(void* const* d_in, const int* in_sizes, int n_in,
                              void* d_out, int out_size, void* d_ws, size_t ws_size,
                              hipStream_t stream) {
    const float* x    = (const float*)d_in[0];
    const float* Wa   = (const float*)d_in[1];
    const float* Ua   = (const float*)d_in[2];
    const float* ba   = (const float*)d_in[3];
    const float* Va   = (const float*)d_in[4];
    const float* Wih  = (const float*)d_in[5];
    const float* Whh  = (const float*)d_in[6];
    const float* bih  = (const float*)d_in[7];
    const float* bhh  = (const float*)d_in[8];
    const float* fcw  = (const float*)d_in[9];
    const float* fcb  = (const float*)d_in[10];
    float* out = (float*)d_out;

    char* ws = (char*)d_ws;
    unsigned short* gfrag  = (unsigned short*)(ws + OFF_GFRAG);
    unsigned short* ufrag  = (unsigned short*)(ws + OFF_UFRAG);
    unsigned short* wafrag = (unsigned short*)(ws + OFF_WAFRAG);
    unsigned short* vafrag = (unsigned short*)(ws + OFF_VAFRAG);
    unsigned short* xatt   = (unsigned short*)(ws + OFF_XATT);
    unsigned short* hbf    = (unsigned short*)(ws + OFF_HBF);
    unsigned short* cbf    = (unsigned short*)(ws + OFF_CBF);
    float*          hf     = (float*)(ws + OFF_HF);
    float*          cf     = (float*)(ws + OFF_CF);
    float*          gates  = (float*)(ws + OFF_GATES);

    prep_weights<<<(2448 * 512 + 255) / 256, 256, 0, stream>>>(
        Wa, Ua, Va, Wih, Whh, gfrag, ufrag, wafrag, vafrag);
    init_state<<<(BATCH * HID + 255) / 256, 256, 0, stream>>>(hf, cf, hbf, cbf);

    for (int t = 0; t < SEQ; ++t) {
        attn_step<<<BATCH / 16, 128, 0, stream>>>(x, ba, wafrag, ufrag, vafrag,
                                                  hbf, cbf, xatt, t);
        gates_gemm<<<128, 256, 0, stream>>>(xatt, hbf, gfrag, gates);
        cell_update<<<(BATCH * HID + 255) / 256, 256, 0, stream>>>(
            gates, bih, bhh, hf, cf, hbf, cbf);
    }
    final_fc<<<(BATCH * PRED + 255) / 256, 256, 0, stream>>>(hf, fcw, fcb, out);
}
